// TransformerXLModel_30296699306621
// MI455X (gfx1250) — compile-verified, matrix operand order checked by reading, not run on hardware
//
#include <hip/hip_runtime.h>
#include <stdint.h>

// ---------------- problem constants ----------------
#define Lc 4
#define Hc 1024
#define Nc 16
#define Dc 64
#define Fc 4096
#define Bc 2
#define Qc 512
#define Mc 512
#define Cc (Qc + Mc)      // 1024
#define Rc (Cc + Qc)      // 1536
#define LN_EPS 1e-12f
#define NEG_INF_F (-1e30f)

typedef __attribute__((ext_vector_type(16))) __bf16 v16bf;
typedef __attribute__((ext_vector_type(8)))  float  v8f;

// ---------------- bf16 helpers ----------------
__device__ __forceinline__ uint16_t f2bf(float f) {
  union { float f; uint32_t u; } c; c.f = f;
  uint32_t u = c.u;
  uint32_t r = u + 0x7FFFu + ((u >> 16) & 1u);   // round-to-nearest-even
  if ((u & 0x7F800000u) == 0x7F800000u) r = u;   // inf/nan passthrough
  return (uint16_t)(r >> 16);
}

// ================= WMMA bf16 GEMM =================
// C[M,N] = A[M,K] * B   (bf16 inputs, fp32 out)
// REQUIRES: M,N multiples of 64; K multiple of 32; lda/ldb multiples of 8;
//           all base pointers 16-byte aligned. (True for every call here.)
// transB==0: B is K x N row-major.  transB==1: B is N x K row-major (C=A*B^T).
// Two-level batching: grid.z = outer*inner, per-level pointer strides.
#define TM 64
#define TN 64
#define TK 32
#define LDP 48   // padded LDS row stride in bf16 elems; 48*2=96B, 16B-aligned rows

__global__ __launch_bounds__(256)
void gemm_bf16_kernel(const uint16_t* __restrict__ A,
                      const uint16_t* __restrict__ Bm,
                      float* __restrict__ C,
                      int K,
                      int lda, int ldb, int ldc,
                      int transB, int inner_cnt,
                      long long sa_o, long long sa_i,
                      long long sb_o, long long sb_i,
                      long long sc_o, long long sc_i)
{
  __shared__ __align__(16) uint16_t sA[TM * LDP];
  __shared__ __align__(16) uint16_t sB[TN * LDP];

  const int z  = blockIdx.z;
  const int zo = z / inner_cnt;
  const int zi = z % inner_cnt;
  A  += zo * sa_o + zi * sa_i;
  Bm += zo * sb_o + zi * sb_i;
  C  += zo * sc_o + zi * sc_i;

  const int m0   = blockIdx.y * TM;
  const int n0   = blockIdx.x * TN;
  const int tid  = threadIdx.x;
  const int wave = tid >> 5;
  const int lane = tid & 31;
  const int trow  = wave & 3;          // tile row 0..3
  const int tcol0 = (wave >> 2) * 2;   // tile cols {0,1} or {2,3}
  const int mloc  = lane & 15;
  const int half  = lane >> 4;

  // staging coordinates (8 bf16 = one uint4 per thread per tile)
  const int ar  = tid >> 2;            // 0..63  row
  const int ac8 = (tid & 3) * 8;       // 0,8,16,24  k-offset
  const int bk  = tid >> 3;            // 0..31  k-row (for transB==0)
  const int bn8 = (tid & 7) * 8;       // 0..56  n-offset (for transB==0)

  v8f acc0 = {};
  v8f acc1 = {};

  for (int k0 = 0; k0 < K; k0 += TK) {
    // ---- stage A tile (64 x 32): vector load + vector LDS store ----
    {
      uint4 va = *(const uint4*)(A + (long long)(m0 + ar) * lda + k0 + ac8);
      *(uint4*)(&sA[ar * LDP + ac8]) = va;
    }
    // ---- stage B tile as [n][k] (64 x 32) ----
    if (transB) {
      uint4 vb = *(const uint4*)(Bm + (long long)(n0 + ar) * ldb + k0 + ac8);
      *(uint4*)(&sB[ar * LDP + ac8]) = vb;
    } else {
      union { uint4 v; uint16_t h[8]; } vb;
      vb.v = *(const uint4*)(Bm + (long long)(k0 + bk) * ldb + n0 + bn8);
#pragma unroll
      for (int e = 0; e < 8; ++e) sB[(bn8 + e) * LDP + bk] = vb.h[e];
    }
    // prefetch next K-slab of A into cache (global_prefetch path)
    if (k0 + TK < K) {
      __builtin_prefetch(A + (long long)(m0 + ar) * lda + k0 + TK + ac8, 0, 1);
    }
    __syncthreads();

    // ---- build fragments per documented CDNA5 16-bit layouts ----
    union { v16bf v; uint32_t u[8]; } fa, fb0, fb1;
    const uint32_t* pA  = (const uint32_t*)&sA[(trow * 16 + mloc) * LDP];
    const uint32_t* pB0 = (const uint32_t*)&sB[((tcol0    ) * 16 + mloc) * LDP];
    const uint32_t* pB1 = (const uint32_t*)&sB[((tcol0 + 1) * 16 + mloc) * LDP];
#pragma unroll
    for (int g = 0; g < 8; ++g) {
      int kk = (g < 4) ? (2 * g + 8 * half) : (2 * (g - 4) + 8 * half + 16);
      fa.u[g]  = pA [kk >> 1];
      fb0.u[g] = pB0[kk >> 1];
      fb1.u[g] = pB1[kk >> 1];
    }
    acc0 = __builtin_amdgcn_wmma_f32_16x16x32_bf16(false, fa.v, false, fb0.v,
                                                   (short)0, acc0, false, false);
    acc1 = __builtin_amdgcn_wmma_f32_16x16x32_bf16(false, fa.v, false, fb1.v,
                                                   (short)0, acc1, false, false);
    __syncthreads();
  }

  // ---- writeback (C/D layout: VGPR g -> M = g + 8*half, N = lane&15) ----
  const int cn0 = n0 + tcol0 * 16 + mloc;
  const int cn1 = cn0 + 16;
#pragma unroll
  for (int g = 0; g < 8; ++g) {
    int gm = m0 + trow * 16 + g + 8 * half;
    long long off = (long long)gm * ldc;
    C[off + cn0] = acc0[g];
    C[off + cn1] = acc1[g];
  }
}

// ================= elementwise / reduction kernels =================

__global__ void cvt_bf16_kernel(const float* __restrict__ in,
                                uint16_t* __restrict__ out, long long n) {
  long long idx = (long long)blockIdx.x * blockDim.x + threadIdx.x;
  if (idx < n) out[idx] = f2bf(in[idx]);
}

__global__ void build_ctx_kernel(const float* __restrict__ mems_l,
                                 const float* __restrict__ x,
                                 uint16_t* __restrict__ ctx, long long total) {
  long long idx = (long long)blockIdx.x * blockDim.x + threadIdx.x;
  if (idx >= total) return;
  int h = (int)(idx % Hc);
  long long t = idx / Hc;
  int j = (int)(t % Cc);
  int b = (int)(t / Cc);
  float v = (j < Mc) ? mems_l[((long long)b * Mc + j) * Hc + h]
                     : x[((long long)b * Qc + (j - Mc)) * Hc + h];
  ctx[idx] = f2bf(v);
}

__global__ void qbias_kernel(const float* __restrict__ q,
                             const float* __restrict__ cb,
                             const float* __restrict__ pb,
                             uint16_t* __restrict__ qcb,
                             uint16_t* __restrict__ qpb, long long total) {
  long long idx = (long long)blockIdx.x * blockDim.x + threadIdx.x;
  if (idx >= total) return;
  int c = (int)(idx % (Nc * Dc));
  float qq = q[idx];
  qcb[idx] = f2bf(qq + cb[c]);
  qpb[idx] = f2bf(qq + pb[c]);
}

__global__ void ef2_kernel(const float* __restrict__ q,
                           const float* __restrict__ sb,
                           const float* __restrict__ se,   // seg_enc[l]: [2,N,D]
                           float* __restrict__ out, int total) {
  int idx = blockIdx.x * blockDim.x + threadIdx.x;
  if (idx >= total) return;
  int s = idx & 1;
  int t = idx >> 1;            // (b*N + n)*Q + i
  int i = t % Qc;
  int bn = t / Qc;
  int n = bn % Nc;
  int b = bn / Nc;
  const float* qr  = q  + ((long long)b * Qc + i) * Hc + n * Dc;
  const float* sbn = sb + n * Dc;
  const float* sen = se + ((long long)s * Nc + n) * Dc;
  float acc = 0.f;
#pragma unroll 8
  for (int d = 0; d < Dc; ++d) acc += (qr[d] + sbn[d]) * sen[d];
  out[idx] = acc;
}

// fused: scores = (ac + relshift(bd) + ef)*scale + mask*NEG_INF ; softmax ; bf16
__global__ __launch_bounds__(256)
void attn_softmax_kernel(const float* __restrict__ ac,          // [B*N*Q, C]
                         const float* __restrict__ bd,          // [B*N*Q, R]
                         const float* __restrict__ ef2,         // [B*N*Q, 2]
                         const unsigned char* __restrict__ seg, // [B,Q,C] bool
                         const float* __restrict__ mask,        // [B,1,Q,C]
                         uint16_t* __restrict__ probs, float scale) {
  __shared__ float red[256];
  const int row = blockIdx.x;            // (b*N + n)*Q + i
  const int i  = row % Qc;
  const int bn = row / Qc;
  const int b  = bn / Nc;
  const float* acr = ac + (long long)row * Cc;
  const float* bdr = bd + (long long)row * Rc;
  const float e0 = ef2[row * 2 + 0];
  const float e1 = ef2[row * 2 + 1];
  const unsigned char* segr = seg + ((long long)b * Qc + i) * Cc;
  const float* maskr = mask + ((long long)b * Qc + i) * Cc;
  const int tid = threadIdx.x;

  float sc[Cc / 256];
  float lmax = -3.4e38f;
#pragma unroll
  for (int t = 0; t < Cc / 256; ++t) {
    int j = tid + t * 256;
    float s = acr[j] + bdr[Qc + j - i] + (segr[j] ? e1 : e0);
    s = s * scale + maskr[j] * NEG_INF_F;
    sc[t] = s;
    lmax = fmaxf(lmax, s);
  }
  red[tid] = lmax; __syncthreads();
  for (int sft = 128; sft > 0; sft >>= 1) {
    if (tid < sft) red[tid] = fmaxf(red[tid], red[tid + sft]);
    __syncthreads();
  }
  const float m = red[0]; __syncthreads();
  float lsum = 0.f;
#pragma unroll
  for (int t = 0; t < Cc / 256; ++t) { sc[t] = __expf(sc[t] - m); lsum += sc[t]; }
  red[tid] = lsum; __syncthreads();
  for (int sft = 128; sft > 0; sft >>= 1) {
    if (tid < sft) red[tid] += red[tid + sft];
    __syncthreads();
  }
  const float inv = 1.f / red[0];
  uint16_t* pr = probs + (long long)row * Cc;
#pragma unroll
  for (int t = 0; t < Cc / 256; ++t) pr[tid + t * 256] = f2bf(sc[t] * inv);
}

// out = LN(main + res? + bias?) * gamma + beta ; optional bf16 mirror
__global__ __launch_bounds__(256)
void resid_ln_kernel(const float* __restrict__ m_,
                     const float* __restrict__ r_,
                     const float* __restrict__ bias,
                     const float* __restrict__ g_,
                     const float* __restrict__ b_,
                     float* __restrict__ of, uint16_t* __restrict__ ob) {
  __shared__ float s1[256], s2[256];
  const long long row = blockIdx.x;
  const float* mr = m_ + row * Hc;
  const float* rr = r_ ? r_ + row * Hc : nullptr;
  const int tid = threadIdx.x;
  float v[Hc / 256];
  float sum = 0.f, sq = 0.f;
#pragma unroll
  for (int t = 0; t < Hc / 256; ++t) {
    int h = tid + 256 * t;
    float x = mr[h];
    if (rr)   x += rr[h];
    if (bias) x += bias[h];
    v[t] = x; sum += x; sq += x * x;
  }
  s1[tid] = sum; s2[tid] = sq; __syncthreads();
  for (int sft = 128; sft > 0; sft >>= 1) {
    if (tid < sft) { s1[tid] += s1[tid + sft]; s2[tid] += s2[tid + sft]; }
    __syncthreads();
  }
  const float mu  = s1[0] * (1.f / Hc);
  const float var = s2[0] * (1.f / Hc) - mu * mu;
  const float inv = rsqrtf(var + LN_EPS);
#pragma unroll
  for (int t = 0; t < Hc / 256; ++t) {
    int h = tid + 256 * t;
    float o = (v[t] - mu) * inv * g_[h] + b_[h];
    of[row * Hc + h] = o;
    if (ob) ob[row * Hc + h] = f2bf(o);
  }
}

__global__ void bias_relu_kernel(const float* __restrict__ in,
                                 const float* __restrict__ bias,
                                 uint16_t* __restrict__ ob,
                                 long long total, int cols) {
  long long idx = (long long)blockIdx.x * blockDim.x + threadIdx.x;
  if (idx >= total) return;
  float v = in[idx] + bias[idx % cols];
  ob[idx] = f2bf(fmaxf(v, 0.f));
}

// ================= host launcher =================
typedef long long ll;

static void gemm(hipStream_t stream,
                 const uint16_t* A, const uint16_t* Bm, float* C,
                 int M, int N, int K, int lda, int ldb, int ldc,
                 int transB, int outer, int inner,
                 ll sao, ll sai, ll sbo, ll sbi, ll sco, ll sci) {
  dim3 g((N + TN - 1) / TN, (M + TM - 1) / TM, outer * inner);
  gemm_bf16_kernel<<<g, 256, 0, stream>>>(A, Bm, C, K, lda, ldb, ldc,
                                          transB, inner,
                                          sao, sai, sbo, sbi, sco, sci);
}

extern "C" void kernel_launch(void* const* d_in, const int* in_sizes, int n_in,
                              void* d_out, int out_size, void* d_ws, size_t ws_size,
                              hipStream_t stream) {
  const float* content_stream = (const float*)d_in[0];
  const float* content_mask   = (const float*)d_in[1];
  const float* pos_enc        = (const float*)d_in[2];
  const unsigned char* segmat = (const unsigned char*)d_in[3];
  const float* mems           = (const float*)d_in[4];
  const float* w_q            = (const float*)d_in[5];
  const float* w_k            = (const float*)d_in[6];
  const float* w_v            = (const float*)d_in[7];
  const float* w_r            = (const float*)d_in[8];
  const float* w_o            = (const float*)d_in[9];
  const float* ffn_w1         = (const float*)d_in[10];
  const float* ffn_b1         = (const float*)d_in[11];
  const float* ffn_w2         = (const float*)d_in[12];
  const float* ffn_b2         = (const float*)d_in[13];
  const float* ln1_g          = (const float*)d_in[14];
  const float* ln1_b          = (const float*)d_in[15];
  const float* ln2_g          = (const float*)d_in[16];
  const float* ln2_b          = (const float*)d_in[17];
  const float* content_bias   = (const float*)d_in[18];
  const float* position_bias  = (const float*)d_in[19];
  const float* segment_bias   = (const float*)d_in[20];
  const float* segment_enc    = (const float*)d_in[21];

  // ---- workspace bump allocator ----
  char* wp = (char*)d_ws;
  auto alloc = [&](size_t bytes) -> void* {
    void* p = (void*)wp;
    wp += (bytes + 255) & ~(size_t)255;
    return p;
  };
  float*    x_buf   = (float*)   alloc(sizeof(float) * Bc * Qc * Hc);
  float*    y_buf   = (float*)   alloc(sizeof(float) * Bc * Qc * Hc);
  uint16_t* ctx_bf  = (uint16_t*)alloc(sizeof(uint16_t) * Bc * Cc * Hc);
  uint16_t* pos_bf  = (uint16_t*)alloc(sizeof(uint16_t) * Bc * Rc * Hc);
  float*    q_f     = (float*)   alloc(sizeof(float) * Bc * Qc * Hc);
  float*    k_f     = (float*)   alloc(sizeof(float) * Bc * Cc * Hc);
  float*    v_f     = (float*)   alloc(sizeof(float) * Bc * Cc * Hc);
  float*    r_f     = (float*)   alloc(sizeof(float) * Bc * Rc * Hc);
  uint16_t* k_bf    = (uint16_t*)alloc(sizeof(uint16_t) * Bc * Cc * Hc);
  uint16_t* v_bf    = (uint16_t*)alloc(sizeof(uint16_t) * Bc * Cc * Hc);
  uint16_t* r_bf    = (uint16_t*)alloc(sizeof(uint16_t) * Bc * Rc * Hc);
  uint16_t* qcb_bf  = (uint16_t*)alloc(sizeof(uint16_t) * Bc * Qc * Hc);
  uint16_t* qpb_bf  = (uint16_t*)alloc(sizeof(uint16_t) * Bc * Qc * Hc);
  float*    ef2_f   = (float*)   alloc(sizeof(float) * Bc * Nc * Qc * 2);
  float*    scores  = (float*)   alloc(sizeof(float) * (ll)Bc * Nc * Qc * Cc);
  float*    bd_buf  = (float*)   alloc(sizeof(float) * (ll)Bc * Nc * Qc * Rc);
  uint16_t* probs   = (uint16_t*)alloc(sizeof(uint16_t) * (ll)Bc * Nc * Qc * Cc);
  float*    attn_f  = (float*)   alloc(sizeof(float) * Bc * Qc * Hc);
  uint16_t* attn_bf = (uint16_t*)alloc(sizeof(uint16_t) * Bc * Qc * Hc);
  float*    ao_f    = (float*)   alloc(sizeof(float) * Bc * Qc * Hc);
  uint16_t* y_bf    = (uint16_t*)alloc(sizeof(uint16_t) * Bc * Qc * Hc);
  float*    hid_f   = (float*)   alloc(sizeof(float) * (ll)Bc * Qc * Fc);
  uint16_t* hid_bf  = (uint16_t*)alloc(sizeof(uint16_t) * (ll)Bc * Qc * Fc);
  float*    ffo_f   = (float*)   alloc(sizeof(float) * Bc * Qc * Hc);
  uint16_t* wq_bf   = (uint16_t*)alloc(sizeof(uint16_t) * Hc * Hc);
  uint16_t* wk_bf   = (uint16_t*)alloc(sizeof(uint16_t) * Hc * Hc);
  uint16_t* wv_bf   = (uint16_t*)alloc(sizeof(uint16_t) * Hc * Hc);
  uint16_t* wr_bf   = (uint16_t*)alloc(sizeof(uint16_t) * Hc * Hc);
  uint16_t* wo_bf   = (uint16_t*)alloc(sizeof(uint16_t) * Hc * Hc);
  uint16_t* w1_bf   = (uint16_t*)alloc(sizeof(uint16_t) * (ll)Hc * Fc);
  uint16_t* w2_bf   = (uint16_t*)alloc(sizeof(uint16_t) * (ll)Fc * Hc);
  (void)ws_size; (void)n_in; (void)in_sizes; (void)out_size;

  auto cvt = [&](const float* src, uint16_t* dst, ll n) {
    cvt_bf16_kernel<<<(unsigned)((n + 255) / 256), 256, 0, stream>>>(src, dst, n);
  };

  // x = content_stream ; pos_bf = bf16(position_encoding)
  hipMemcpyAsync(x_buf, content_stream, sizeof(float) * Bc * Qc * Hc,
                 hipMemcpyDeviceToDevice, stream);
  cvt(pos_enc, pos_bf, (ll)Bc * Rc * Hc);

  const float scale = 0.125f;  // D^-0.5

  for (int l = 0; l < Lc; ++l) {
    // context = concat(mems[l], x) -> bf16
    build_ctx_kernel<<<(unsigned)(((ll)Bc * Cc * Hc + 255) / 256), 256, 0, stream>>>(
        mems + (ll)l * Bc * Mc * Hc, x_buf, ctx_bf, (ll)Bc * Cc * Hc);

    // per-layer weight conversion to bf16
    cvt(w_q + (ll)l * Hc * Hc, wq_bf, (ll)Hc * Hc);
    cvt(w_k + (ll)l * Hc * Hc, wk_bf, (ll)Hc * Hc);
    cvt(w_v + (ll)l * Hc * Hc, wv_bf, (ll)Hc * Hc);
    cvt(w_r + (ll)l * Hc * Hc, wr_bf, (ll)Hc * Hc);
    cvt(w_o + (ll)l * Hc * Hc, wo_bf, (ll)Hc * Hc);
    cvt(ffn_w1 + (ll)l * Hc * Fc, w1_bf, (ll)Hc * Fc);
    cvt(ffn_w2 + (ll)l * Fc * Hc, w2_bf, (ll)Fc * Hc);

    // projections (NN GEMMs, batch over B)
    gemm(stream, ctx_bf + (ll)Mc * Hc, wq_bf, q_f, Qc, Hc, Hc, Hc, Hc, Hc, 0,
         Bc, 1, (ll)Cc * Hc, 0, 0, 0, (ll)Qc * Hc, 0);
    gemm(stream, ctx_bf, wk_bf, k_f, Cc, Hc, Hc, Hc, Hc, Hc, 0,
         Bc, 1, (ll)Cc * Hc, 0, 0, 0, (ll)Cc * Hc, 0);
    gemm(stream, ctx_bf, wv_bf, v_f, Cc, Hc, Hc, Hc, Hc, Hc, 0,
         Bc, 1, (ll)Cc * Hc, 0, 0, 0, (ll)Cc * Hc, 0);
    gemm(stream, pos_bf, wr_bf, r_f, Rc, Hc, Hc, Hc, Hc, Hc, 0,
         Bc, 1, (ll)Rc * Hc, 0, 0, 0, (ll)Rc * Hc, 0);

    cvt(k_f, k_bf, (ll)Bc * Cc * Hc);
    cvt(v_f, v_bf, (ll)Bc * Cc * Hc);
    cvt(r_f, r_bf, (ll)Bc * Rc * Hc);

    qbias_kernel<<<(unsigned)(((ll)Bc * Qc * Hc + 255) / 256), 256, 0, stream>>>(
        q_f, content_bias, position_bias, qcb_bf, qpb_bf, (ll)Bc * Qc * Hc);

    // ac[b,n] = (q+cb) * k^T   (NT, batch over B*N)
    gemm(stream, qcb_bf, k_bf, scores, Qc, Cc, Dc, Hc, Hc, Cc, 1,
         Bc, Nc, (ll)Qc * Hc, Dc, (ll)Cc * Hc, Dc,
         (ll)Nc * Qc * Cc, (ll)Qc * Cc);
    // bd[b,n] = (q+pb) * r^T   (NT, batch over B*N)
    gemm(stream, qpb_bf, r_bf, bd_buf, Qc, Rc, Dc, Hc, Hc, Rc, 1,
         Bc, Nc, (ll)Qc * Hc, Dc, (ll)Rc * Hc, Dc,
         (ll)Nc * Qc * Rc, (ll)Qc * Rc);

    ef2_kernel<<<(Bc * Nc * Qc * 2 + 255) / 256, 256, 0, stream>>>(
        q_f, segment_bias, segment_enc + (ll)l * 2 * Nc * Dc, ef2_f,
        Bc * Nc * Qc * 2);

    attn_softmax_kernel<<<Bc * Nc * Qc, 256, 0, stream>>>(
        scores, bd_buf, ef2_f, segmat, content_mask, probs, scale);

    // attn[b,n] = probs * v    (NN, batch over B*N)
    gemm(stream, probs, v_bf, attn_f, Qc, Dc, Cc, Cc, Hc, Hc, 0,
         Bc, Nc, (ll)Nc * Qc * Cc, (ll)Qc * Cc, (ll)Cc * Hc, Dc,
         (ll)Qc * Hc, Dc);
    cvt(attn_f, attn_bf, (ll)Bc * Qc * Hc);

    // attn_out = attn * w_o^T  (NT, batch over B)
    gemm(stream, attn_bf, wo_bf, ao_f, Qc, Hc, Hc, Hc, Hc, Hc, 1,
         Bc, 1, (ll)Qc * Hc, 0, 0, 0, (ll)Qc * Hc, 0);

    // y = LN1(attn_out + x)
    resid_ln_kernel<<<Bc * Qc, 256, 0, stream>>>(
        ao_f, x_buf, nullptr, ln1_g + (ll)l * Hc, ln1_b + (ll)l * Hc,
        y_buf, y_bf);

    // FFN
    gemm(stream, y_bf, w1_bf, hid_f, Qc, Fc, Hc, Hc, Fc, Fc, 0,
         Bc, 1, (ll)Qc * Hc, 0, 0, 0, (ll)Qc * Fc, 0);
    bias_relu_kernel<<<(unsigned)(((ll)Bc * Qc * Fc + 255) / 256), 256, 0, stream>>>(
        hid_f, ffn_b1 + (ll)l * Fc, hid_bf, (ll)Bc * Qc * Fc, Fc);
    gemm(stream, hid_bf, w2_bf, ffo_f, Qc, Hc, Fc, Fc, Hc, Hc, 0,
         Bc, 1, (ll)Qc * Fc, 0, 0, 0, (ll)Qc * Hc, 0);

    // x = LN2(ffn_out + b2 + y)
    resid_ln_kernel<<<Bc * Qc, 256, 0, stream>>>(
        ffo_f, y_buf, ffn_b2 + (ll)l * Hc, ln2_g + (ll)l * Hc,
        ln2_b + (ll)l * Hc, x_buf, nullptr);
  }

  hipMemcpyAsync(d_out, x_buf, sizeof(float) * Bc * Qc * Hc,
                 hipMemcpyDeviceToDevice, stream);
}
